// LocalTransformer_70652212019541
// MI455X (gfx1250) — matrix operand order repaired and checked by reading, not compiled, not run on hardware
//
#include <hip/hip_runtime.h>

typedef __bf16 bf16;
typedef __attribute__((ext_vector_type(16))) __bf16 v16bf;
typedef __attribute__((ext_vector_type(8)))  float  v8f;
typedef __attribute__((ext_vector_type(4)))  unsigned int u32x4;
typedef __attribute__((ext_vector_type(8)))  int i32x8;
typedef __attribute__((ext_vector_type(4)))  int i32x4;

#define WMMA_BF16(a,b,c) __builtin_amdgcn_wmma_f32_16x16x32_bf16(false,(a),false,(b),(short)0,(c),false,false)

#if defined(__has_builtin)
#if __has_builtin(__builtin_amdgcn_tensor_load_to_lds) && __has_builtin(__builtin_amdgcn_s_wait_tensorcnt)
#define HAS_TDM 1
#endif
#endif
#if __has_include(<hip/amd_detail/amd_gfx1250_TDM.h>)
#define TDM_SIX_ARGS 1
#endif

#ifdef HAS_TDM
// LDS byte offset: generic LDS addresses carry the offset in addr[31:0] (ISA 10.2)
__device__ __forceinline__ unsigned lds_byte_off(const void* p) {
  return (unsigned)(unsigned long long)(uintptr_t)p;
}
// 2-D bf16 tile DMA: tile (tile0 x tile1) elems from row-major tensor (stride0 elems/row),
// OOB (x >= tdim0) reads return zero -> free K-tail padding.
__device__ __forceinline__ void tdm_load_2d(unsigned ldsOff, const void* g,
                                            int tile0, int tile1,
                                            int tdim0, int tdim1, int stride0) {
  unsigned long long ga = (unsigned long long)(uintptr_t)g;
  u32x4 g0;
  g0[0] = 1u;                                                  // count=1, user mode
  g0[1] = ldsOff;                                              // lds_addr
  g0[2] = (unsigned)ga;                                        // global_addr[31:0]
  g0[3] = (unsigned)((ga >> 32) & 0x01FFFFFFu) | 0x80000000u;  // addr[56:32] | type=2
  i32x8 g1;
  g1[0] = 0x00010000;                                // workgroup_mask=0, data_size=1 (2B)
  g1[1] = (tdim0 & 0xFFFF) << 16;                    // tensor_dim0[15:0] @ [63:48]
  g1[2] = ((tdim0 >> 16) & 0xFFFF) | ((tdim1 & 0xFFFF) << 16);  // dim0 hi, dim1 lo
  g1[3] = ((tdim1 >> 16) & 0xFFFF) | ((tile0 & 0xFFFF) << 16);  // dim1 hi, tile_dim0
  g1[4] = tile1 & 0xFFFF;                            // tile_dim1 (tile_dim2=0)
  g1[5] = stride0;                                   // tensor_dim0_stride[31:0]
  g1[6] = 0;                                         // stride0 hi, dim1_stride lo
  g1[7] = 0;
  i32x4 z4; z4[0] = 0; z4[1] = 0; z4[2] = 0; z4[3] = 0;
#ifdef TDM_SIX_ARGS
  i32x8 z8; for (int i = 0; i < 8; ++i) z8[i] = 0;
  __builtin_amdgcn_tensor_load_to_lds(g0, g1, z4, z4, z8, 0);
#else
  __builtin_amdgcn_tensor_load_to_lds(g0, g1, z4, z4, 0);
#endif
}
#endif  // HAS_TDM

// ---------- fragment loaders (CDNA5 WMMA 16x16x32 bf16 layouts) ----------
// A (16x32): lane<16 -> row=lane, K {0..7,16..23}; lane>=16 -> row=lane-16, K {8..15,24..31}
__device__ __forceinline__ v16bf load_fragA(const bf16* p) {
  v16bf f;
#pragma unroll
  for (int i = 0; i < 8; ++i) { f[i] = p[i]; f[i + 8] = p[i + 16]; }
  return f;
}
// B (32x16): lane holds col n = lane%16, K {0..7,16..23} (lane<16) / {8..15,24..31} (lane>=16)
__device__ __forceinline__ v16bf load_fragB(const bf16* p) {
  v16bf f;
#pragma unroll
  for (int i = 0; i < 8; ++i) { f[i] = p[i]; f[i + 8] = p[i + 16]; }
  return f;
}
__device__ __forceinline__ float half16_max(float v) {
#pragma unroll
  for (int m = 1; m < 16; m <<= 1) v = fmaxf(v, __shfl_xor(v, m, 32));
  return v;
}
__device__ __forceinline__ float half16_sum(float v) {
#pragma unroll
  for (int m = 1; m < 16; m <<= 1) v += __shfl_xor(v, m, 32);
  return v;
}

// ---------- weight pack: W (E rows x K cols, f32) -> Bp bf16 [Kpad/32][Npad][32] ----------
__global__ void pack_w_kernel(const float* __restrict__ W, bf16* __restrict__ P,
                              int E, int K, int Npad) {
  const int Kpad = (K + 31) & ~31;
  const int total = Kpad * Npad;
  int idx = blockIdx.x * blockDim.x + threadIdx.x;
  if (idx >= total) return;
  int kb  = idx / (Npad * 32);
  int rem = idx - kb * (Npad * 32);
  int n   = rem >> 5;
  int k   = (kb << 5) + (rem & 31);
  float v = (n < E && k < K) ? W[(size_t)n * K + k] : 0.f;
  P[idx] = (bf16)v;
}

// ---------- tiled WMMA GEMM: C(MxNout) = A(MxK) * Bpacked + (res) ----------
#define BM 128
#define BN 64
#define BK 32
template <bool A_BF16, bool OUT_BF16, bool HAS_RES>
__global__ __launch_bounds__(256) void gemm_wmma_kernel(
    const void* __restrict__ Av, const bf16* __restrict__ Bp, void* __restrict__ Cv,
    const float* __restrict__ Res, int M, int K, int Nout, int Npad) {
#ifdef HAS_TDM
  __shared__ alignas(16) bf16 lds_a[2][BM * BK];
  __shared__ alignas(16) bf16 lds_b[2][BN * BK];
#else
  __shared__ alignas(16) bf16 lds_a1[BM * BK];
  __shared__ alignas(16) bf16 lds_b1[BN * BK];
#endif
  const int tid  = threadIdx.x;
  const int wav  = tid >> 5;
  const int lane = tid & 31;
  const int hl   = lane >> 4;
  const int r    = lane & 15;
  const int m0 = blockIdx.y * BM;
  const int n0 = blockIdx.x * BN;
  const int Kpad = (K + 31) & ~31;

  const float* Af = (const float*)Av;
  const bf16*  Ab = (const bf16*)Av;

  v8f acc[4];
#pragma unroll
  for (int j = 0; j < 4; ++j)
#pragma unroll
    for (int v = 0; v < 8; ++v) acc[j][v] = 0.f;

#ifdef HAS_TDM
  // wave0 issues double-buffered TDM DMA for the LDS tiles
  if (tid < 32) {
    if (A_BF16)
      tdm_load_2d(lds_byte_off(lds_a[0]), Ab + (size_t)m0 * K, BK, BM, K, BM, K);
    tdm_load_2d(lds_byte_off(lds_b[0]), Bp + (size_t)n0 * 32, BN * BK, 1, BN * BK, 1, BN * BK);
  }
#endif

  for (int k0 = 0; k0 < Kpad; k0 += BK) {
#ifdef HAS_TDM
    const int cur = (k0 >> 5) & 1;
    bf16* la = lds_a[cur];
    bf16* lb = lds_b[cur];
    if (k0 + BK < Kpad && tid < 32) {  // prefetch next tiles into other buffer
      if (A_BF16)
        tdm_load_2d(lds_byte_off(lds_a[cur ^ 1]), Ab + (size_t)m0 * K + (k0 + BK),
                    BK, BM, K - (k0 + BK), BM, K);
      tdm_load_2d(lds_byte_off(lds_b[cur ^ 1]),
                  Bp + (size_t)((k0 + BK) >> 5) * Npad * 32 + (size_t)n0 * 32,
                  BN * BK, 1, BN * BK, 1, BN * BK);
    }
    if (!A_BF16) {  // f32 activations: manual convert-stage (TDM cannot convert)
      int row  = tid >> 1;
      int col0 = (tid & 1) * 16;
      size_t g = (size_t)(m0 + row) * K + k0 + col0;
      if (k0 + BK < Kpad) __builtin_prefetch(&Af[g + BK], 0, 0);
#pragma unroll
      for (int i = 0; i < 16; ++i) {
        int kk = k0 + col0 + i;
        la[row * BK + col0 + i] = (bf16)((kk < K) ? Af[g + i] : 0.f);
      }
    }
    if (tid < 32) {  // current tiles' DMA complete (next tiles may be in flight)
      if (k0 + BK < Kpad) __builtin_amdgcn_s_wait_tensorcnt(A_BF16 ? 2 : 1);
      else                __builtin_amdgcn_s_wait_tensorcnt(0);
    }
#else
    bf16* la = lds_a1;
    bf16* lb = lds_b1;
    {
      int row  = tid >> 1;
      int col0 = (tid & 1) * 16;
      size_t g = (size_t)(m0 + row) * K + k0 + col0;
#pragma unroll
      for (int i = 0; i < 16; ++i) {
        int kk = k0 + col0 + i;
        float v = 0.f;
        if (kk < K) v = A_BF16 ? (float)Ab[g + i] : Af[g + i];
        la[row * BK + col0 + i] = (bf16)v;
      }
    }
    {
      const uint4* s4 = (const uint4*)(Bp + (size_t)(k0 >> 5) * Npad * 32 + (size_t)n0 * 32);
      ((uint4*)lb)[tid] = s4[tid];
    }
#endif
    __syncthreads();
    v16bf afrag = load_fragA(&la[(wav * 16 + r) * BK + hl * 8]);
#pragma unroll
    for (int j = 0; j < 4; ++j) {
      v16bf bfrag = load_fragB(&lb[(j * 16 + r) * BK + hl * 8]);
      acc[j] = WMMA_BF16(afrag, bfrag, acc[j]);
    }
    __syncthreads();
  }
  // epilogue: C row = half*8+v, col = lane%16
#pragma unroll
  for (int j = 0; j < 4; ++j)
#pragma unroll
    for (int v = 0; v < 8; ++v) {
      int mr  = m0 + wav * 16 + hl * 8 + v;
      int col = n0 + j * 16 + r;
      if (col < Nout) {
        size_t idx = (size_t)mr * Nout + col;
        float val  = acc[j][v];
        if (HAS_RES) val += Res[idx];
        if (OUT_BF16) ((bf16*)Cv)[idx] = (bf16)val;
        else          ((float*)Cv)[idx] = val;
      }
    }
}

// ---------- local windowed attention (flash-style, window=128, look 1 back/fwd) ----------
__global__ __launch_bounds__(256) void attn_kernel(const bf16* __restrict__ qkv,
                                                   bf16* __restrict__ outp) {
  __shared__ alignas(16) bf16 lds_v[64 * 128];      // [d][key] for current key chunk
  __shared__ alignas(16) bf16 lds_p[8 * 16 * 128];  // per-wave P (A-layout staging)
  const int win = blockIdx.x;   // 0..31
  const int hd  = blockIdx.y;   // 0..7
  const int b   = blockIdx.z;   // 0..7
  const int tid = threadIdx.x;
  const int wav = tid >> 5, lane = tid & 31, hl = lane >> 4, r = lane & 15;
  const int nw = 32;
  const size_t RS = 1536;  // qkv row stride
  const bf16* qbase = qkv + (size_t)b * 4096 * RS + hd * 64;
  const bf16* kbase = qbase + 512;
  const bf16* vbase = qbase + 1024;

  // Q fragments (A operand), rows win*128 + wav*16 + r
  v16bf qf[2];
  {
    const bf16* qp = qbase + (size_t)(win * 128 + wav * 16 + r) * RS;
    qf[0] = load_fragA(qp + 0 * 32 + hl * 8);
    qf[1] = load_fragA(qp + 1 * 32 + hl * 8);
  }
  float mrow[8], lrow[8];
  v8f o[4];
#pragma unroll
  for (int v = 0; v < 8; ++v) { mrow[v] = -3.0e38f; lrow[v] = 0.f; }
#pragma unroll
  for (int t = 0; t < 4; ++t)
#pragma unroll
    for (int v = 0; v < 8; ++v) o[t][v] = 0.f;

  bf16* myP = lds_p + wav * 16 * 128;

  for (int c = 0; c < 3; ++c) {
    int kw = win - 1 + c;
    if (kw < 0 || kw >= nw) continue;  // uniform across the workgroup
    __syncthreads();                   // previous chunk's lds_v readers done
    // stage V chunk transposed: lds_v[d*128 + key]; 4 keys packed per b64 DS store
    {
      int kg = tid >> 3;            // key group 0..31 -> keys kg*4 .. kg*4+3
      int d0 = (tid & 7) * 8;       // d chunk
      const bf16* vp = vbase + (size_t)(kw * 128 + kg * 4) * RS + d0;
      bf16 rw[4][8];
#pragma unroll
      for (int j = 0; j < 4; ++j)
#pragma unroll
        for (int i = 0; i < 8; ++i) rw[j][i] = vp[(size_t)j * RS + i];
#pragma unroll
      for (int i = 0; i < 8; ++i) {
        union { bf16 h[4]; unsigned long long u; } pk;
        pk.h[0] = rw[0][i]; pk.h[1] = rw[1][i]; pk.h[2] = rw[2][i]; pk.h[3] = rw[3][i];
        *(unsigned long long*)&lds_v[(d0 + i) * 128 + kg * 4] = pk.u;
      }
    }
    // S = q @ k^T  (8 n-tiles of 16 keys, K=64 in two steps)
    v8f s[8];
#pragma unroll
    for (int t = 0; t < 8; ++t)
#pragma unroll
      for (int v = 0; v < 8; ++v) s[t][v] = 0.f;
#pragma unroll
    for (int t = 0; t < 8; ++t) {
      const bf16* kp = kbase + (size_t)(kw * 128 + t * 16 + r) * RS;
      v16bf b0 = load_fragB(kp + 0 * 32 + hl * 8);
      s[t] = WMMA_BF16(qf[0], b0, s[t]);
      v16bf b1 = load_fragB(kp + 1 * 32 + hl * 8);
      s[t] = WMMA_BF16(qf[1], b1, s[t]);
    }
    // online softmax update
    const float scale = 0.125f;  // 64^-0.5
    float newm[8];
#pragma unroll
    for (int v = 0; v < 8; ++v) {
      float mx = -3.0e38f;
#pragma unroll
      for (int t = 0; t < 8; ++t) {
        float sv = (float)s[t][v] * scale;
        s[t][v] = sv;
        mx = fmaxf(mx, sv);
      }
      mx      = half16_max(mx);
      newm[v] = fmaxf(mrow[v], mx);
      float alpha = __expf(mrow[v] - newm[v]);
#pragma unroll
      for (int t = 0; t < 4; ++t) o[t][v] *= alpha;
      lrow[v] *= alpha;
      mrow[v] = newm[v];
    }
#pragma unroll
    for (int v = 0; v < 8; ++v) {
      float ls = 0.f;
#pragma unroll
      for (int t = 0; t < 8; ++t) {
        float p = __expf((float)s[t][v] - newm[v]);
        ls += p;
        myP[(hl * 8 + v) * 128 + t * 16 + r] = (bf16)p;  // A-layout row-major
      }
      lrow[v] += half16_sum(ls);
    }
    __syncthreads();  // lds_v (cross-wave) + myP visible
    // O += P @ V  (K=128 keys in 4 steps, 4 d-tiles)
#pragma unroll
    for (int ks = 0; ks < 4; ++ks) {
      v16bf pf = load_fragA(&myP[r * 128 + ks * 32 + hl * 8]);
#pragma unroll
      for (int t2 = 0; t2 < 4; ++t2) {
        v16bf vf = load_fragB(&lds_v[(t2 * 16 + r) * 128 + ks * 32 + hl * 8]);
        o[t2] = WMMA_BF16(pf, vf, o[t2]);
      }
    }
  }
  // normalize and store (B*N x 512 bf16)
#pragma unroll
  for (int t2 = 0; t2 < 4; ++t2)
#pragma unroll
    for (int v = 0; v < 8; ++v) {
      int mr  = b * 4096 + win * 128 + wav * 16 + hl * 8 + v;
      int col = hd * 64 + t2 * 16 + r;
      outp[(size_t)mr * 512 + col] = (bf16)(o[t2][v] / lrow[v]);
    }
}

// ---------- embed: h[m,d] = x[m]*w[d] + b[d] ----------
__global__ void embed_kernel(const float* __restrict__ x, const float* __restrict__ w,
                             const float* __restrict__ bias, float* __restrict__ h) {
  int idx = blockIdx.x * blockDim.x + threadIdx.x;
  if (idx >= 32768 * 512) return;
  int m = idx >> 9, d = idx & 511;
  h[idx] = x[m] * w[d] + bias[d];
}

// ---------- LayerNorm (width 512), one wave per row, f32 in -> bf16 out ----------
__global__ __launch_bounds__(256) void ln_kernel(const float* __restrict__ h,
                                                 const float* __restrict__ g,
                                                 const float* __restrict__ bb,
                                                 bf16* __restrict__ y, int M) {
  int wav = threadIdx.x >> 5, lane = threadIdx.x & 31;
  int row = blockIdx.x * 8 + wav;
  if (row >= M) return;
  const float* x = h + (size_t)row * 512;
  float vals[16], s = 0.f, s2 = 0.f;
#pragma unroll
  for (int i = 0; i < 16; ++i) {
    float v = x[lane * 16 + i];
    vals[i] = v; s += v; s2 += v * v;
  }
#pragma unroll
  for (int m = 1; m < 32; m <<= 1) { s += __shfl_xor(s, m, 32); s2 += __shfl_xor(s2, m, 32); }
  float mu = s * (1.f / 512.f);
  float rs = rsqrtf(s2 * (1.f / 512.f) - mu * mu + 1e-5f);
  bf16* yo = y + (size_t)row * 512;
#pragma unroll
  for (int i = 0; i < 16; ++i) {
    int cidx = lane * 16 + i;
    yo[cidx] = (bf16)((vals[i] - mu) * rs * g[cidx] + bb[cidx]);
  }
}

// ---------- GEGLU: y = u * gelu_exact(gate) ----------
__global__ void geglu_kernel(const bf16* __restrict__ ff1, bf16* __restrict__ y, int total) {
  int idx = blockIdx.x * blockDim.x + threadIdx.x;
  if (idx >= total) return;
  int m = idx / 1365, i = idx - m * 1365;
  float u  = (float)ff1[(size_t)m * 2730 + i];
  float gt = (float)ff1[(size_t)m * 2730 + 1365 + i];
  float ge = 0.5f * gt * (1.f + erff(gt * 0.70710678118f));
  y[idx] = (bf16)(u * ge);
}

// ---------- mean pool over N ----------
__global__ void pool_kernel(const float* __restrict__ h, float* __restrict__ pooled) {
  int t = blockIdx.x * blockDim.x + threadIdx.x;
  if (t >= 8 * 512) return;
  int b = t >> 9, d = t & 511;
  const float* p = h + (size_t)b * 4096 * 512 + d;
  float s = 0.f;
  for (int n = 0; n < 4096; ++n) s += p[(size_t)n * 512];
  pooled[t] = s * (1.f / 4096.f);
}

// ---------- final projection: out[b,c] = pooled[b] . proj_w[c] + pb[c] ----------
__global__ __launch_bounds__(256) void proj_kernel(const float* __restrict__ pooled,
                                                   const float* __restrict__ pw,
                                                   const float* __restrict__ pb,
                                                   float* __restrict__ out) {
  int wav = threadIdx.x >> 5, lane = threadIdx.x & 31;
  int g = blockIdx.x * 8 + wav;
  if (g >= 80) return;
  int b = g / 10, c = g - b * 10;
  const float* pp = pooled + b * 512;
  const float* ww = pw + c * 512;
  float s = 0.f;
#pragma unroll
  for (int i = 0; i < 16; ++i) s += pp[lane * 16 + i] * ww[lane * 16 + i];
#pragma unroll
  for (int m = 1; m < 32; m <<= 1) s += __shfl_xor(s, m, 32);
  if (lane == 0) out[b * 10 + c] = s + pb[c];
}

// =======================================================================
extern "C" void kernel_launch(void* const* d_in, const int* in_sizes, int n_in,
                              void* d_out, int out_size, void* d_ws, size_t ws_size,
                              hipStream_t stream) {
  (void)in_sizes; (void)n_in; (void)out_size; (void)ws_size;
  const float* x_in   = (const float*)d_in[0];
  const float* emb_w  = (const float*)d_in[1];
  const float* emb_b  = (const float*)d_in[2];
  const float* qkv_w  = (const float*)d_in[3];
  const float* out_w  = (const float*)d_in[4];
  const float* ln_g   = (const float*)d_in[5];
  const float* ln_b   = (const float*)d_in[6];
  const float* ff_w1  = (const float*)d_in[7];
  const float* ff_w2  = (const float*)d_in[8];
  const float* proj_w = (const float*)d_in[9];
  const float* proj_b = (const float*)d_in[10];
  float* out = (float*)d_out;

  const int M = 32768;  // B*N
  char* ws = (char*)d_ws;
  const size_t MBy = 1ull << 20;
  float* h      = (float*)(ws + 0);                 // 64 MB f32
  char*  regA   = ws + 64 * MBy;                    // 180 MB: qkv bf16 (96MB) then ff1out bf16 (171MB)
  char*  regB   = regA + 180 * MBy;                 // 92 MB: attnout | lny | geglu (sequential lifetimes)
  bf16*  qkvbuf = (bf16*)regA;
  bf16*  ff1out = (bf16*)regA;
  bf16*  attnout= (bf16*)regB;
  bf16*  lny    = (bf16*)regB;
  bf16*  geglu  = (bf16*)regB;
  char*  pk     = regB + 92 * MBy;
  bf16*  qkvP   = (bf16*)pk;                        // 512*1536*2
  bf16*  outP   = (bf16*)(pk + 1572864);            // 512*512*2
  bf16*  ff1P   = (bf16*)(pk + 1572864 + 524288);   // 512*2752*2
  bf16*  ff2P   = (bf16*)(pk + 1572864 + 524288 + 2818048);  // 1376*512*2
  float* pooled = (float*)(pk + 1572864 + 524288 + 2818048 + 1409024);

  // embedding
  embed_kernel<<<(32768 * 512) / 256, 256, 0, stream>>>(x_in, emb_w, emb_b, h);

  for (int l = 0; l < 4; ++l) {
    // pack this layer's weights into WMMA B-operand layout
    pack_w_kernel<<<(512 * 1536 + 255) / 256, 256, 0, stream>>>(
        qkv_w + (size_t)l * 1536 * 512, qkvP, 1536, 512, 1536);
    pack_w_kernel<<<(512 * 512 + 255) / 256, 256, 0, stream>>>(
        out_w + (size_t)l * 512 * 512, outP, 512, 512, 512);
    pack_w_kernel<<<(512 * 2752 + 255) / 256, 256, 0, stream>>>(
        ff_w1 + (size_t)l * 2730 * 512, ff1P, 2730, 512, 2752);
    pack_w_kernel<<<(1376 * 512 + 255) / 256, 256, 0, stream>>>(
        ff_w2 + (size_t)l * 512 * 1365, ff2P, 512, 1365, 512);

    // qkv = h @ Wqkv^T   (f32 A -> bf16 out)
    gemm_wmma_kernel<false, true, false><<<dim3(1536 / 64, M / 128), 256, 0, stream>>>(
        h, qkvP, qkvbuf, nullptr, M, 512, 1536, 1536);
    // local windowed attention
    attn_kernel<<<dim3(32, 8, 8), 256, 0, stream>>>(qkvbuf, attnout);
    // h += attnout @ Wout^T
    gemm_wmma_kernel<true, false, true><<<dim3(512 / 64, M / 128), 256, 0, stream>>>(
        attnout, outP, h, h, M, 512, 512, 512);
    // y = LN(h)
    ln_kernel<<<M / 8, 256, 0, stream>>>(h, ln_g + l * 512, ln_b + l * 512, lny, M);
    // ff1 = y @ W1^T
    gemm_wmma_kernel<true, true, false><<<dim3(2752 / 64, M / 128), 256, 0, stream>>>(
        lny, ff1P, ff1out, nullptr, M, 512, 2730, 2752);
    // geglu
    geglu_kernel<<<(M * 1365 + 255) / 256, 256, 0, stream>>>(ff1out, geglu, M * 1365);
    // h += geglu @ W2^T
    gemm_wmma_kernel<true, false, true><<<dim3(512 / 64, M / 128), 256, 0, stream>>>(
        geglu, ff2P, h, h, M, 1365, 512, 512);
  }

  pool_kernel<<<(8 * 512) / 256, 256, 0, stream>>>(h, pooled);
  proj_kernel<<<10, 256, 0, stream>>>(pooled, proj_w, proj_b, out);
}
